// ORGAN_67834713473538
// MI455X (gfx1250) — matrix operand order repaired and checked by reading, not compiled, not run on hardware
//
#include <hip/hip_runtime.h>

#define HID    512
#define H4     2048
#define EMB    32
#define VOC    30
#define BATCH  1024
#define TSTEPS 64
#define PAD_T  0
#define EOS_T  2

typedef __attribute__((ext_vector_type(8)))  __bf16 v8bf;
typedef __attribute__((ext_vector_type(16))) __bf16 v16bf;
typedef __attribute__((ext_vector_type(8)))  float  v8f;

__device__ __forceinline__ v16bf organ_cat8(v8bf lo, v8bf hi) {
    return __builtin_shufflevector(lo, hi, 0,1,2,3,4,5,6,7,8,9,10,11,12,13,14,15);
}

__device__ __forceinline__ float organ_sigm(float x) {
    return 1.0f / (1.0f + __expf(-x));
}

// ---------------------------------------------------------------------------
// f32 -> bf16 weight conversion
// ---------------------------------------------------------------------------
__global__ void organ_cvt_bf16(const float* __restrict__ src,
                               __bf16* __restrict__ dst, int n) {
    int i = blockIdx.x * blockDim.x + threadIdx.x;
    if (i < n) dst[i] = (__bf16)src[i];
}

// ---------------------------------------------------------------------------
// Per-launch state init (graph-replay determinism)
// ---------------------------------------------------------------------------
__global__ void organ_init(const int* __restrict__ prevs_in,
                           float* __restrict__ cA, float* __restrict__ cB,
                           __bf16* __restrict__ hA0, __bf16* __restrict__ hB0,
                           int* __restrict__ prev, int* __restrict__ is_end,
                           int* __restrict__ lengths) {
    int i = blockIdx.x * blockDim.x + threadIdx.x;
    if (i < BATCH * HID) {
        cA[i] = 0.0f; cB[i] = 0.0f;
        hA0[i] = (__bf16)0.0f; hB0[i] = (__bf16)0.0f;
    }
    if (i < BATCH) {
        int p = prevs_in[i];
        prev[i] = p;
        is_end[i] = (p == EOS_T) ? 1 : 0;
        lengths[i] = 0;
    }
}

// ---------------------------------------------------------------------------
// x = emb[prev]  (gather, f32 -> bf16)
// ---------------------------------------------------------------------------
__global__ void organ_embed(const float* __restrict__ emb,
                            const int* __restrict__ prev,
                            __bf16* __restrict__ xbf) {
    int i = blockIdx.x * blockDim.x + threadIdx.x;   // B*EMB threads
    int b = i >> 5, e = i & 31;
    xbf[i] = (__bf16)emb[prev[b] * EMB + e];
}

// ---------------------------------------------------------------------------
// Accumulate  acc += X[m0..m0+15, :] @ W[gate*512 + j0 .. +15, :]^T
// Weight slice (64 rows x K-chunk) is staged through LDS once per block and
// shared by all 8 waves (8x cut in L2 weight traffic). Padded pitch avoids
// LDS bank conflicts on the 16-consecutive-row fragment reads.
// ---------------------------------------------------------------------------
#define KCHUNK 256
#define SPAD   8                          // pad (bf16 elems) -> +16B per row

__device__ __forceinline__ void organ_accum_part(
    const __bf16* __restrict__ Xg,        // [B, Kw] activations (global)
    const __bf16* __restrict__ Wg,        // [2048, Kw] weights (global)
    int Kw, int j0, int m0, int lane,
    __bf16* sB, v8f acc[4])
{
    const int tid  = threadIdx.x;
    const int l15  = lane & 15;
    const int half = lane >> 4;
    const int kbA  = half << 3;           // A frag K sub-offset {0,8}
    const int kbB  = half << 3;           // B frag K sub-offset {0,8}
    const __bf16* arow = Xg + (size_t)(m0 + l15) * Kw;

    for (int kc0 = 0; kc0 < Kw; kc0 += KCHUNK) {
        const int KC    = (Kw - kc0 < KCHUNK) ? (Kw - kc0) : KCHUNK;  // 256 or 32
        const int pitch = KC + SPAD;
        const int kvs   = (KC == KCHUNK) ? 5 : 2;   // log2(uint4 per row)
        const int nvec  = 64 << kvs;

        __syncthreads();
        // ---- stage 64 gate-rows x KC weights into LDS (uint4 copies) ----
        for (int i = tid; i < nvec; i += 256) {
            int rl = i >> kvs;
            int kv = i & ((1 << kvs) - 1);
            int n  = ((rl >> 4) * HID) + j0 + (rl & 15);
            *(uint4*)(sB + rl * pitch + (kv << 3)) =
                *(const uint4*)(Wg + (size_t)n * Kw + kc0 + (kv << 3));
        }
        __syncthreads();

        // ---- consume: A from global(L2), B from LDS ----
        for (int kc = 0; kc < KC; kc += 32) {
            v16bf a = organ_cat8(*(const v8bf*)(arow + kc0 + kc + kbA),
                                 *(const v8bf*)(arow + kc0 + kc + kbA + 16));
            #pragma unroll
            for (int g = 0; g < 4; ++g) {
                const __bf16* bp = sB + ((g << 4) + l15) * pitch + kc + kbB;
                v16bf bfrag = organ_cat8(*(const v8bf*)bp, *(const v8bf*)(bp + 16));
                acc[g] = __builtin_amdgcn_wmma_f32_16x16x32_bf16(
                    false, a, false, bfrag, (short)0, acc[g], false, false);
            }
        }
    }
}

// ---------------------------------------------------------------------------
// Generic LSTM cell: G[B,2048] = X[B,Kx]@Wih^T + H[B,512]@Whh^T + bih + bhh
// then c' = sig(f)*c + sig(i)*tanh(g);  hout = sig(o)*tanh(c')
// Grid: 256 blocks = (j_tile 0..31) x (m_group 0..7); 8 waves/block, each
// wave owns one 16-row batch tile; all waves share the block's weight slice.
// ---------------------------------------------------------------------------
__global__ __launch_bounds__(256) void organ_lstm_cell(
    const __bf16* __restrict__ X, int Kx,
    const __bf16* __restrict__ Wih,        // [2048, Kx] row-major bf16
    const __bf16* __restrict__ H,          // [B, 512] bf16
    const __bf16* __restrict__ Whh,        // [2048, 512] row-major bf16
    const float*  __restrict__ bih, const float* __restrict__ bhh,
    float* __restrict__ C,                 // [B, 512] f32, in-place
    __bf16* __restrict__ Hout)             // [B, 512] bf16
{
    __shared__ __bf16 sB[64 * (KCHUNK + SPAD)];   // 33.8 KB

    const int lane    = threadIdx.x & 31;
    const int wave    = threadIdx.x >> 5;
    const int j_tile  = blockIdx.x & 31;
    const int m_group = blockIdx.x >> 5;
    const int j0      = j_tile << 4;
    const int m0      = (m_group << 7) + (wave << 4);
    const int l15     = lane & 15;
    const int half    = lane >> 4;

    v8f acc[4];
    #pragma unroll
    for (int g = 0; g < 4; ++g) acc[g] = (v8f){0,0,0,0,0,0,0,0};

    organ_accum_part(X, Wih, Kx,  j0, m0, lane, sB, acc);   // x @ Wih^T
    organ_accum_part(H, Whh, HID, j0, m0, lane, sB, acc);   // h @ Whh^T

    // ---- epilogue: lane holds (M = r + half*8, N = l15) for r = 0..7 ----
    const int j = j0 + l15;                // hidden unit index 0..511
    float bI = bih[0*HID + j] + bhh[0*HID + j];
    float bF = bih[1*HID + j] + bhh[1*HID + j];
    float bG = bih[2*HID + j] + bhh[2*HID + j];
    float bO = bih[3*HID + j] + bhh[3*HID + j];

    #pragma unroll
    for (int r = 0; r < 8; ++r) {
        int m = m0 + r + (half << 3);
        size_t idx = (size_t)m * HID + j;
        float gi = organ_sigm(acc[0][r] + bI);
        float gf = organ_sigm(acc[1][r] + bF);
        float gg = tanhf(acc[2][r] + bG);
        float go = organ_sigm(acc[3][r] + bO);
        float cn = gf * C[idx] + gi * gg;
        C[idx] = cn;
        Hout[idx] = (__bf16)(go * tanhf(cn));
    }
}

// ---------------------------------------------------------------------------
// logits = hB @ w_out^T + b_out; Gumbel-max sample; state update; outputs
// one thread per batch row (30x512 dots — tiny VALU work, w_out uniform)
// ---------------------------------------------------------------------------
__global__ __launch_bounds__(256) void organ_sample(
    const __bf16* __restrict__ Hb, const float* __restrict__ wout,
    const float* __restrict__ bout, int t,
    int* __restrict__ prev, int* __restrict__ is_end, int* __restrict__ lengths,
    float* __restrict__ seq_out, float* __restrict__ logits_out,
    float* __restrict__ len_out)
{
    int b = blockIdx.x * blockDim.x + threadIdx.x;
    if (b >= BATCH) return;

    float acc[VOC];
    #pragma unroll
    for (int v = 0; v < VOC; ++v) acc[v] = 0.0f;

    const __bf16* h = Hb + (size_t)b * HID;
    for (int k = 0; k < HID; ++k) {
        float hv = (float)h[k];
        #pragma unroll
        for (int v = 0; v < VOC; ++v) acc[v] = fmaf(hv, wout[v * HID + k], acc[v]);
    }

    // Gumbel-max categorical sampling with a deterministic hash PRNG
    unsigned int seed = (unsigned int)(t + 1) * 0x9E3779B9u ^ (unsigned int)b * 0x85EBCA77u;
    float best = -1e30f;
    int arg = 0;
    #pragma unroll
    for (int v = 0; v < VOC; ++v) {
        float logit = acc[v] + bout[v];
        logits_out[((size_t)b * TSTEPS + t) * VOC + v] = logit;
        unsigned int x = seed ^ ((unsigned int)v * 0x6C8E9CF5u);
        x ^= x >> 16; x *= 0x7FEB352Du;
        x ^= x >> 15; x *= 0x846CA68Bu;
        x ^= x >> 16;
        float u = ((float)(x >> 8) + 0.5f) * (1.0f / 16777216.0f);
        float g = -__logf(-__logf(u));
        float val = logit + g;
        if (val > best) { best = val; arg = v; }
    }

    int ended = is_end[b];
    int cur = ended ? PAD_T : arg;
    int len = lengths[b] + (ended ? 0 : 1);
    lengths[b] = len;
    is_end[b] = ended | (cur == EOS_T);
    prev[b] = cur;

    seq_out[(size_t)b * TSTEPS + t] = (float)cur;
    len_out[b] = (float)len;
}

// ---------------------------------------------------------------------------
// host launcher
// ---------------------------------------------------------------------------
extern "C" void kernel_launch(void* const* d_in, const int* in_sizes, int n_in,
                              void* d_out, int out_size, void* d_ws, size_t ws_size,
                              hipStream_t stream) {
    const int*   prevs  = (const int*)  d_in[0];
    const float* emb    = (const float*)d_in[1];
    const float* w_ih_0 = (const float*)d_in[2];
    const float* w_hh_0 = (const float*)d_in[3];
    const float* b_ih_0 = (const float*)d_in[4];
    const float* b_hh_0 = (const float*)d_in[5];
    const float* w_ih_1 = (const float*)d_in[6];
    const float* w_hh_1 = (const float*)d_in[7];
    const float* b_ih_1 = (const float*)d_in[8];
    const float* b_hh_1 = (const float*)d_in[9];
    const float* w_out  = (const float*)d_in[10];
    const float* b_out  = (const float*)d_in[11];

    // workspace layout (256B-aligned slabs)
    char* ws = (char*)d_ws;
    size_t off = 0;
    auto carve = [&](size_t bytes) {
        char* p = ws + off;
        off += (bytes + 255) & ~(size_t)255;
        return p;
    };
    __bf16* wih0 = (__bf16*)carve((size_t)H4 * EMB * 2);
    __bf16* whh0 = (__bf16*)carve((size_t)H4 * HID * 2);
    __bf16* wih1 = (__bf16*)carve((size_t)H4 * HID * 2);
    __bf16* whh1 = (__bf16*)carve((size_t)H4 * HID * 2);
    __bf16* xbf  = (__bf16*)carve((size_t)BATCH * EMB * 2);
    __bf16* hA[2] = { (__bf16*)carve((size_t)BATCH * HID * 2),
                      (__bf16*)carve((size_t)BATCH * HID * 2) };
    __bf16* hB[2] = { (__bf16*)carve((size_t)BATCH * HID * 2),
                      (__bf16*)carve((size_t)BATCH * HID * 2) };
    float* cA = (float*)carve((size_t)BATCH * HID * 4);
    float* cB = (float*)carve((size_t)BATCH * HID * 4);
    int* prev    = (int*)carve(BATCH * 4);
    int* is_end  = (int*)carve(BATCH * 4);
    int* lengths = (int*)carve(BATCH * 4);

    float* seq_out    = (float*)d_out;                              // [B,64]
    float* logits_out = seq_out + (size_t)BATCH * TSTEPS;           // [B,64,30]
    float* len_out    = logits_out + (size_t)BATCH * TSTEPS * VOC;  // [B]

    // weight conversion (every launch; deterministic)
    {
        int n0 = H4 * EMB, n1 = H4 * HID;
        organ_cvt_bf16<<<(n0 + 255) / 256, 256, 0, stream>>>(w_ih_0, wih0, n0);
        organ_cvt_bf16<<<(n1 + 255) / 256, 256, 0, stream>>>(w_hh_0, whh0, n1);
        organ_cvt_bf16<<<(n1 + 255) / 256, 256, 0, stream>>>(w_ih_1, wih1, n1);
        organ_cvt_bf16<<<(n1 + 255) / 256, 256, 0, stream>>>(w_hh_1, whh1, n1);
    }
    organ_init<<<(BATCH * HID + 255) / 256, 256, 0, stream>>>(
        prevs, cA, cB, hA[0], hB[0], prev, is_end, lengths);

    for (int t = 0; t < TSTEPS; ++t) {
        int pi = t & 1, po = pi ^ 1;
        organ_embed<<<(BATCH * EMB) / 256, 256, 0, stream>>>(emb, prev, xbf);
        organ_lstm_cell<<<256, 256, 0, stream>>>(
            xbf, EMB, wih0, hA[pi], whh0, b_ih_0, b_hh_0, cA, hA[po]);
        organ_lstm_cell<<<256, 256, 0, stream>>>(
            hA[po], HID, wih1, hB[pi], whh1, b_ih_1, b_hh_1, cB, hB[po]);
        organ_sample<<<BATCH / 256, 256, 0, stream>>>(
            hB[po], w_out, b_out, t, prev, is_end, lengths,
            seq_out, logits_out, len_out);
    }
}
